// Transformer_82867099009859
// MI455X (gfx1250) — matrix operand order repaired and checked by reading, not compiled, not run on hardware
//
#include <hip/hip_runtime.h>

// ---------------------------------------------------------------------------
// Transformer encoder-decoder forward pass for MI455X (gfx1250).
// All matmuls (QKV/O, attention scores, P*V, FFN, logits) run on
// V_WMMA_F32_16X16X32_BF16 with fp32 accumulation. Global<->LDS staging uses
// 128-bit accesses (float4) with load/store batching to avoid per-element
// waitcnt serialization.
// ---------------------------------------------------------------------------

#define D_MODEL 512
#define NHEAD   8
#define DKH     64          // D_MODEL / NHEAD
#define FF_DIM  2048
#define VOCAB   10000
#define NLAYER  6
#define BATCH   8
#define SEQ     512
#define NTOK    (BATCH * SEQ)
#define LN_EPS  1e-5f

typedef __attribute__((ext_vector_type(16))) __bf16 v16bf;
typedef __attribute__((ext_vector_type(8)))  float  v8f;
typedef __attribute__((ext_vector_type(4)))  float  f4;

static __device__ __forceinline__ v8f wmma_bf16(v16bf a, v16bf b, v8f c) {
    // D = A(16x32 bf16) * B(32x16 bf16) + C(16x16 f32)
    return __builtin_amdgcn_wmma_f32_16x16x32_bf16(
        false, a, false, b, (short)0, c, false, false);
}

// K index inside a 16x32 bf16 A fragment for vector element e, lane-half lh.
// Contiguous order (matches B fills): lanes 0-15 -> K {0..15}, lanes 16-31 -> K {16..31}.
static __device__ __forceinline__ int a_kidx(int e, int lh) {
    return lh * 16 + e;
}

// ---------------------------------------------------------------------------
// GEMM: C[M,N] = A[M,K] @ W[K,N] + bias[N], optional ReLU.
// Block: 256 threads (8 waves). Block tile 64x64, K-step 32.
// Wave (wm,wn) computes rows wm*16..+15, cols wn*32..+31 (two 16x16 accums).
// M, K must be multiples of 64/32 (true for all call sites); N is guarded at
// float4 granularity (all call-site N are multiples of 4).
// ---------------------------------------------------------------------------
__global__ __launch_bounds__(256)
void gemm_bias_kernel(const float* __restrict__ A, const float* __restrict__ W,
                      const float* __restrict__ bias, float* __restrict__ C,
                      int M, int N, int K, int relu) {
    __shared__ __align__(16) float lsA[64 * 32];
    __shared__ __align__(16) float lsB[32 * 64];

    const int tid  = threadIdx.x;
    const int lane = tid & 31;
    const int wave = tid >> 5;
    const int wm   = wave & 3;       // M subtile (4 x 16 rows)
    const int wn   = wave >> 2;      // N subtile (2 x 32 cols)
    const int lh   = lane >> 4;
    const int lm   = lane & 15;
    const int mBase = blockIdx.y * 64;
    const int nBase = blockIdx.x * 64;

    // staging coordinates (128-bit granules)
    const int arow = tid >> 3;           // 0..31  (A: 8 float4 per 32-float row)
    const int acol = (tid & 7) * 4;
    const int brow = tid >> 4;           // 0..15  (B: 16 float4 per 64-float row)
    const int bcol = (tid & 15) * 4;
    const int nb   = nBase + bcol;

    v8f acc0 = {}, acc1 = {};

    for (int k0 = 0; k0 < K; k0 += 32) {
        // ---- issue all global loads into registers first ----
        f4 a0 = *(const f4*)(A + (size_t)(mBase + arow)      * K + k0 + acol);
        f4 a1 = *(const f4*)(A + (size_t)(mBase + 32 + arow) * K + k0 + acol);
        f4 b0v = {0.f, 0.f, 0.f, 0.f};
        f4 b1v = {0.f, 0.f, 0.f, 0.f};
        if (nb < N) {
            b0v = *(const f4*)(W + (size_t)(k0 + brow)      * N + nb);
            b1v = *(const f4*)(W + (size_t)(k0 + brow + 16) * N + nb);
        }
        __syncthreads();                 // previous iteration's readers done
        *(f4*)&lsA[arow * 32 + acol]        = a0;
        *(f4*)&lsA[(arow + 32) * 32 + acol] = a1;
        *(f4*)&lsB[brow * 64 + bcol]        = b0v;
        *(f4*)&lsB[(brow + 16) * 64 + bcol] = b1v;
        __syncthreads();

        // ---- fragments + WMMA ----
        v16bf af;
        const int ar = wm * 16 + lm;
        #pragma unroll
        for (int e = 0; e < 16; ++e)
            af[e] = (__bf16)lsA[ar * 32 + a_kidx(e, lh)];

        v16bf b0, b1;
        #pragma unroll
        for (int e = 0; e < 16; ++e) {
            int kk = lh * 16 + e;                 // B: lane-half selects K group
            b0[e] = (__bf16)lsB[kk * 64 + wn * 32 + lm];
            b1[e] = (__bf16)lsB[kk * 64 + wn * 32 + 16 + lm];
        }

        acc0 = wmma_bf16(af, b0, acc0);
        acc1 = wmma_bf16(af, b1, acc1);
    }

    // C layout: VGPR r holds M = r + 8*laneHalf, N = lane&15
    #pragma unroll
    for (int r = 0; r < 8; ++r) {
        int m  = mBase + wm * 16 + r + lh * 8;
        int n0 = nBase + wn * 32 + lm;
        int n1 = n0 + 16;
        if (n0 < N) {
            float o = acc0[r] + bias[n0];
            if (relu) o = fmaxf(o, 0.0f);
            C[(size_t)m * N + n0] = o;
        }
        if (n1 < N) {
            float o = acc1[r] + bias[n1];
            if (relu) o = fmaxf(o, 0.0f);
            C[(size_t)m * N + n1] = o;
        }
    }
}

// ---------------------------------------------------------------------------
// Attention core: out = softmax(mask(Q K^T / sqrt(DK))) @ V  per (b, h).
// One wave (32 threads) per 16-query tile; full 16x512 score row in LDS.
// K and V tiles are staged to LDS with float4 loads before fragment gather.
// Q/K/V/O are [B,S,D_MODEL] with head h occupying columns h*64..+63.
// causal==1 -> look-ahead mask; else pad mask from tok[b, key] != 0.
// ---------------------------------------------------------------------------
__global__ __launch_bounds__(32)
void attention_kernel(const float* __restrict__ Q, const float* __restrict__ Kb,
                      const float* __restrict__ Vb, float* __restrict__ O,
                      const int* __restrict__ tok, int causal) {
    __shared__ float sc[16][SEQ + 1];
    __shared__ __align__(16) float kTile[16 * 64];   // 16 keys x 64 head dims
    __shared__ __align__(16) float vTile[32 * 64];   // 32 keys x 64 head dims

    const int bid  = blockIdx.x;
    const int qt   = bid & (SEQ / 16 - 1);      // 32 query tiles
    const int h    = (bid >> 5) & (NHEAD - 1);
    const int b    = bid >> 8;
    const int lane = threadIdx.x;
    const int lh   = lane >> 4;
    const int lm   = lane & 15;
    const int hoff = h * DKH;

    // Q fragments: 16 queries x 64 dims = two K-steps of 32 (one-time load)
    v16bf aq0, aq1;
    {
        size_t base = ((size_t)b * SEQ + qt * 16 + lm) * D_MODEL + hoff;
        #pragma unroll
        for (int e = 0; e < 16; ++e) {
            int k = a_kidx(e, lh);
            aq0[e] = (__bf16)Q[base + k];
            aq1[e] = (__bf16)Q[base + 32 + k];
        }
    }

    // ---- scores: S = Q K^T / 8, masked, into LDS ----
    for (int kt = 0; kt < SEQ / 16; ++kt) {
        // stage K tile: 16x64 f32 = 256 float4, 8 per lane (coalesced)
        f4 tk[8];
        #pragma unroll
        for (int i = 0; i < 8; ++i) {
            int j = i * 32 + lane;
            int row = j >> 4, col = (j & 15) * 4;
            tk[i] = *(const f4*)(Kb + ((size_t)b * SEQ + kt * 16 + row) * D_MODEL + hoff + col);
        }
        __syncthreads();
        #pragma unroll
        for (int i = 0; i < 8; ++i) {
            int j = i * 32 + lane;
            int row = j >> 4, col = (j & 15) * 4;
            *(f4*)&kTile[row * 64 + col] = tk[i];
        }
        __syncthreads();

        v16bf bk0, bk1;                       // B = K^T: contraction = head dim
        #pragma unroll
        for (int e = 0; e < 16; ++e) {
            int d = lh * 16 + e;
            bk0[e] = (__bf16)kTile[lm * 64 + d];
            bk1[e] = (__bf16)kTile[lm * 64 + 32 + d];
        }
        v8f accs = {};
        accs = wmma_bf16(aq0, bk0, accs);
        accs = wmma_bf16(aq1, bk1, accs);
        #pragma unroll
        for (int r = 0; r < 8; ++r) {
            int qrow = qt * 16 + r + lh * 8;
            int key  = kt * 16 + lm;
            bool ok  = causal ? (key <= qrow) : (tok[b * SEQ + key] != 0);
            sc[r + lh * 8][key] = ok ? (accs[r] * 0.125f) : -__builtin_inff();
        }
    }
    __syncthreads();

    // ---- row-wise softmax over 512 keys (wave32 shuffles) ----
    for (int r = 0; r < 16; ++r) {
        float mx = -__builtin_inff();
        #pragma unroll
        for (int i = 0; i < SEQ / 32; ++i) mx = fmaxf(mx, sc[r][lane + 32 * i]);
        #pragma unroll
        for (int off = 16; off >= 1; off >>= 1)
            mx = fmaxf(mx, __shfl_xor(mx, off, 32));
        float sum = 0.0f;
        #pragma unroll
        for (int i = 0; i < SEQ / 32; ++i)
            sum += __expf(sc[r][lane + 32 * i] - mx);
        #pragma unroll
        for (int off = 16; off >= 1; off >>= 1)
            sum += __shfl_xor(sum, off, 32);
        float inv = 1.0f / sum;
        #pragma unroll
        for (int i = 0; i < SEQ / 32; ++i) {
            int j = lane + 32 * i;
            sc[r][j] = __expf(sc[r][j] - mx) * inv;
        }
    }

    // ---- out = P @ V : 16 queries x 64 head dims, K over 512 keys ----
    v8f o0 = {}, o1 = {}, o2 = {}, o3 = {};
    for (int ks = 0; ks < SEQ / 32; ++ks) {
        // stage V tile: 32x64 f32 = 512 float4, 16 per lane (coalesced)
        f4 tv[16];
        #pragma unroll
        for (int i = 0; i < 16; ++i) {
            int j = i * 32 + lane;
            int row = j >> 4, col = (j & 15) * 4;
            tv[i] = *(const f4*)(Vb + ((size_t)b * SEQ + ks * 32 + row) * D_MODEL + hoff + col);
        }
        __syncthreads();
        #pragma unroll
        for (int i = 0; i < 16; ++i) {
            int j = i * 32 + lane;
            int row = j >> 4, col = (j & 15) * 4;
            *(f4*)&vTile[row * 64 + col] = tv[i];
        }
        __syncthreads();

        v16bf ap;
        #pragma unroll
        for (int e = 0; e < 16; ++e)
            ap[e] = (__bf16)sc[lm][ks * 32 + a_kidx(e, lh)];

        v16bf bv0, bv1, bv2, bv3;
        #pragma unroll
        for (int e = 0; e < 16; ++e) {
            int krow = (lh * 16 + e) * 64;
            bv0[e] = (__bf16)vTile[krow + lm];
            bv1[e] = (__bf16)vTile[krow + 16 + lm];
            bv2[e] = (__bf16)vTile[krow + 32 + lm];
            bv3[e] = (__bf16)vTile[krow + 48 + lm];
        }
        o0 = wmma_bf16(ap, bv0, o0);
        o1 = wmma_bf16(ap, bv1, o1);
        o2 = wmma_bf16(ap, bv2, o2);
        o3 = wmma_bf16(ap, bv3, o3);
    }
    #pragma unroll
    for (int r = 0; r < 8; ++r) {
        size_t obase = ((size_t)b * SEQ + qt * 16 + r + lh * 8) * D_MODEL + hoff;
        O[obase + lm]      = o0[r];
        O[obase + 16 + lm] = o1[r];
        O[obase + 32 + lm] = o2[r];
        O[obase + 48 + lm] = o3[r];
    }
}

// ---------------------------------------------------------------------------
// Fused residual add + LayerNorm over D_MODEL (one row per block, in-place ok)
// ---------------------------------------------------------------------------
__global__ __launch_bounds__(256)
void add_ln_kernel(const float* __restrict__ x, const float* __restrict__ res,
                   const float* __restrict__ gamma, const float* __restrict__ beta,
                   float* __restrict__ out) {
    __shared__ float s_sum[256], s_sq[256];
    const int row = blockIdx.x, tid = threadIdx.x;
    const size_t base = (size_t)row * D_MODEL;
    float t0 = x[base + tid]       + res[base + tid];
    float t1 = x[base + tid + 256] + res[base + tid + 256];
    s_sum[tid] = t0 + t1;
    s_sq[tid]  = t0 * t0 + t1 * t1;
    __syncthreads();
    for (int off = 128; off >= 1; off >>= 1) {
        if (tid < off) { s_sum[tid] += s_sum[tid + off]; s_sq[tid] += s_sq[tid + off]; }
        __syncthreads();
    }
    float mu  = s_sum[0] * (1.0f / D_MODEL);
    float var = s_sq[0] * (1.0f / D_MODEL) - mu * mu;
    float inv = rsqrtf(var + LN_EPS);
    out[base + tid]       = (t0 - mu) * inv * gamma[tid]       + beta[tid];
    out[base + tid + 256] = (t1 - mu) * inv * gamma[tid + 256] + beta[tid + 256];
}

// ---------------------------------------------------------------------------
// Token embedding + sinusoidal positional encoding (computed on the fly)
// ---------------------------------------------------------------------------
__global__ __launch_bounds__(256)
void embed_kernel(const int* __restrict__ tokens, const float* __restrict__ emb,
                  float* __restrict__ out) {
    const int row = blockIdx.x;             // b*SEQ + s
    const int s   = row & (SEQ - 1);
    const int tid = threadIdx.x;
    const int tok = tokens[row];
    #pragma unroll
    for (int j = 0; j < 2; ++j) {
        int d = tid + 256 * j;
        int i = d >> 1;
        // div = exp(2i * (-ln(10000)/D))
        float div = __expf((float)(2 * i) * (-9.210340371976184f / (float)D_MODEL));
        float ang = (float)s * div;
        float pe  = (d & 1) ? __cosf(ang) : __sinf(ang);
        out[(size_t)row * D_MODEL + d] = emb[(size_t)tok * D_MODEL + d] + pe;
    }
}

// ---------------------------------------------------------------------------
extern "C" void kernel_launch(void* const* d_in, const int* in_sizes, int n_in,
                              void* d_out, int out_size, void* d_ws, size_t ws_size,
                              hipStream_t stream) {
    (void)in_sizes; (void)n_in; (void)out_size; (void)ws_size;

    const int*   src        = (const int*)  d_in[0];
    const int*   tgt        = (const int*)  d_in[1];
    const float* src_emb    = (const float*)d_in[2];
    const float* tgt_emb    = (const float*)d_in[3];
    const float* enc_qkvo_w = (const float*)d_in[4];
    const float* enc_qkvo_b = (const float*)d_in[5];
    const float* enc_ff1_w  = (const float*)d_in[6];
    const float* enc_ff1_b  = (const float*)d_in[7];
    const float* enc_ff2_w  = (const float*)d_in[8];
    const float* enc_ff2_b  = (const float*)d_in[9];
    const float* enc_ln_g   = (const float*)d_in[10];
    const float* enc_ln_bp  = (const float*)d_in[11];
    const float* dec_sa_w   = (const float*)d_in[12];
    const float* dec_sa_b   = (const float*)d_in[13];
    const float* dec_ca_w   = (const float*)d_in[14];
    const float* dec_ca_b   = (const float*)d_in[15];
    const float* dec_ff1_w  = (const float*)d_in[16];
    const float* dec_ff1_b  = (const float*)d_in[17];
    const float* dec_ff2_w  = (const float*)d_in[18];
    const float* dec_ff2_b  = (const float*)d_in[19];
    const float* dec_ln_g   = (const float*)d_in[20];
    const float* dec_ln_bp  = (const float*)d_in[21];
    const float* out_w      = (const float*)d_in[22];
    const float* out_b      = (const float*)d_in[23];

    // Workspace layout (floats): 7 * [NTOK,D] + [NTOK,FF]  ~= 88 MB
    float* ws     = (float*)d_ws;
    const size_t TD = (size_t)NTOK * D_MODEL;
    float* xbuf   = ws;            // encoder activations
    float* qbuf   = xbuf  + TD;    // also reused for o-proj / ffn output
    float* kbuf   = qbuf  + TD;
    float* vbuf   = kbuf  + TD;
    float* abuf   = vbuf  + TD;    // attention-core output
    float* encout = abuf  + TD;    // encoder output (kept for cross-attn)
    float* ybuf   = encout + TD;   // decoder activations
    float* ffbuf  = ybuf  + TD;    // [NTOK, FF_DIM]

    const size_t DD = (size_t)D_MODEL * D_MODEL;
    dim3 blk(256);
    dim3 gemmDD(D_MODEL / 64, NTOK / 64);
    dim3 gemmDF(FF_DIM / 64,  NTOK / 64);
    dim3 gemmOUT((VOCAB + 63) / 64, NTOK / 64);
    const int attnBlocks = BATCH * NHEAD * (SEQ / 16);

    // ------------------------- encoder -------------------------
    embed_kernel<<<NTOK, 256, 0, stream>>>(src, src_emb, xbuf);
    for (int l = 0; l < NLAYER; ++l) {
        const float* w  = enc_qkvo_w + (size_t)l * 4 * DD;
        const float* bi = enc_qkvo_b + (size_t)l * 4 * D_MODEL;
        gemm_bias_kernel<<<gemmDD, blk, 0, stream>>>(xbuf, w + 0 * DD, bi + 0 * D_MODEL, qbuf, NTOK, D_MODEL, D_MODEL, 0);
        gemm_bias_kernel<<<gemmDD, blk, 0, stream>>>(xbuf, w + 1 * DD, bi + 1 * D_MODEL, kbuf, NTOK, D_MODEL, D_MODEL, 0);
        gemm_bias_kernel<<<gemmDD, blk, 0, stream>>>(xbuf, w + 2 * DD, bi + 2 * D_MODEL, vbuf, NTOK, D_MODEL, D_MODEL, 0);
        attention_kernel<<<attnBlocks, 32, 0, stream>>>(qbuf, kbuf, vbuf, abuf, src, 0);
        gemm_bias_kernel<<<gemmDD, blk, 0, stream>>>(abuf, w + 3 * DD, bi + 3 * D_MODEL, qbuf, NTOK, D_MODEL, D_MODEL, 0);
        add_ln_kernel<<<NTOK, 256, 0, stream>>>(xbuf, qbuf,
            enc_ln_g + (size_t)(l * 2 + 0) * D_MODEL, enc_ln_bp + (size_t)(l * 2 + 0) * D_MODEL, xbuf);
        gemm_bias_kernel<<<gemmDF, blk, 0, stream>>>(xbuf, enc_ff1_w + (size_t)l * D_MODEL * FF_DIM,
            enc_ff1_b + (size_t)l * FF_DIM, ffbuf, NTOK, FF_DIM, D_MODEL, 1);
        gemm_bias_kernel<<<gemmDD, blk, 0, stream>>>(ffbuf, enc_ff2_w + (size_t)l * FF_DIM * D_MODEL,
            enc_ff2_b + (size_t)l * D_MODEL, qbuf, NTOK, D_MODEL, FF_DIM, 0);
        add_ln_kernel<<<NTOK, 256, 0, stream>>>(xbuf, qbuf,
            enc_ln_g + (size_t)(l * 2 + 1) * D_MODEL, enc_ln_bp + (size_t)(l * 2 + 1) * D_MODEL, xbuf);
    }
    hipMemcpyAsync(encout, xbuf, TD * sizeof(float), hipMemcpyDeviceToDevice, stream);

    // ------------------------- decoder -------------------------
    embed_kernel<<<NTOK, 256, 0, stream>>>(tgt, tgt_emb, ybuf);
    for (int l = 0; l < NLAYER; ++l) {
        // masked self-attention
        const float* wsa = dec_sa_w + (size_t)l * 4 * DD;
        const float* bsa = dec_sa_b + (size_t)l * 4 * D_MODEL;
        gemm_bias_kernel<<<gemmDD, blk, 0, stream>>>(ybuf, wsa + 0 * DD, bsa + 0 * D_MODEL, qbuf, NTOK, D_MODEL, D_MODEL, 0);
        gemm_bias_kernel<<<gemmDD, blk, 0, stream>>>(ybuf, wsa + 1 * DD, bsa + 1 * D_MODEL, kbuf, NTOK, D_MODEL, D_MODEL, 0);
        gemm_bias_kernel<<<gemmDD, blk, 0, stream>>>(ybuf, wsa + 2 * DD, bsa + 2 * D_MODEL, vbuf, NTOK, D_MODEL, D_MODEL, 0);
        attention_kernel<<<attnBlocks, 32, 0, stream>>>(qbuf, kbuf, vbuf, abuf, tgt, 1);
        gemm_bias_kernel<<<gemmDD, blk, 0, stream>>>(abuf, wsa + 3 * DD, bsa + 3 * D_MODEL, qbuf, NTOK, D_MODEL, D_MODEL, 0);
        add_ln_kernel<<<NTOK, 256, 0, stream>>>(ybuf, qbuf,
            dec_ln_g + (size_t)(l * 3 + 0) * D_MODEL, dec_ln_bp + (size_t)(l * 3 + 0) * D_MODEL, ybuf);

        // cross-attention (K/V from encoder output; pad mask from tgt tokens, as in reference)
        const float* wca = dec_ca_w + (size_t)l * 4 * DD;
        const float* bca = dec_ca_b + (size_t)l * 4 * D_MODEL;
        gemm_bias_kernel<<<gemmDD, blk, 0, stream>>>(ybuf,   wca + 0 * DD, bca + 0 * D_MODEL, qbuf, NTOK, D_MODEL, D_MODEL, 0);
        gemm_bias_kernel<<<gemmDD, blk, 0, stream>>>(encout, wca + 1 * DD, bca + 1 * D_MODEL, kbuf, NTOK, D_MODEL, D_MODEL, 0);
        gemm_bias_kernel<<<gemmDD, blk, 0, stream>>>(encout, wca + 2 * DD, bca + 2 * D_MODEL, vbuf, NTOK, D_MODEL, D_MODEL, 0);
        attention_kernel<<<attnBlocks, 32, 0, stream>>>(qbuf, kbuf, vbuf, abuf, tgt, 0);
        gemm_bias_kernel<<<gemmDD, blk, 0, stream>>>(abuf, wca + 3 * DD, bca + 3 * D_MODEL, qbuf, NTOK, D_MODEL, D_MODEL, 0);
        add_ln_kernel<<<NTOK, 256, 0, stream>>>(ybuf, qbuf,
            dec_ln_g + (size_t)(l * 3 + 1) * D_MODEL, dec_ln_bp + (size_t)(l * 3 + 1) * D_MODEL, ybuf);

        // FFN
        gemm_bias_kernel<<<gemmDF, blk, 0, stream>>>(ybuf, dec_ff1_w + (size_t)l * D_MODEL * FF_DIM,
            dec_ff1_b + (size_t)l * FF_DIM, ffbuf, NTOK, FF_DIM, D_MODEL, 1);
        gemm_bias_kernel<<<gemmDD, blk, 0, stream>>>(ffbuf, dec_ff2_w + (size_t)l * FF_DIM * D_MODEL,
            dec_ff2_b + (size_t)l * D_MODEL, qbuf, NTOK, D_MODEL, FF_DIM, 0);
        add_ln_kernel<<<NTOK, 256, 0, stream>>>(ybuf, qbuf,
            dec_ln_g + (size_t)(l * 3 + 2) * D_MODEL, dec_ln_bp + (size_t)(l * 3 + 2) * D_MODEL, ybuf);
    }

    // ------------------------- output projection -------------------------
    gemm_bias_kernel<<<gemmOUT, blk, 0, stream>>>(ybuf, out_w, out_b,
        (float*)d_out, NTOK, VOCAB, D_MODEL, 0);
}
